// TransformerBlock_28286654611537
// MI455X (gfx1250) — compile-verified, matrix operand order checked by reading, not run on hardware
//
#include <hip/hip_runtime.h>
#include <hip/hip_bf16.h>
#include <cstdint>
#include <cstddef>

// ---------------------------------------------------------------------------
// MI455X (gfx1250) transformer block. wave32, WMMA f32_16x16x32_f16 everywhere.
// fp32 weights stream from HBM once, converted to f16 in registers (HBM floor:
// ~1.3GB @ 23.3TB/s ~ 57us; f16 WMMA compute ~ same scale -> balanced design).
// CDNA5-specific paths: v_wmma, TDM gather (tensor_load_to_lds) for MoE token
// gather, global_load_async_to_lds_b128 for attention Q staging.
// ---------------------------------------------------------------------------

typedef _Float16 half_t;
typedef __attribute__((ext_vector_type(16))) _Float16 v16h;
typedef __attribute__((ext_vector_type(8)))  float    v8f;

#define SEQ    2048
#define DMODEL 2048
#define NHQ    16
#define NHKV   8
#define DHEAD  128
#define NEXP   8
#define DFFN   5632
#define KVDIM  (NHKV*DHEAD)   // 1024
#define EPSV   1e-6f

#if defined(__has_builtin)
#if __has_builtin(__builtin_amdgcn_tensor_load_to_lds) && \
    __has_builtin(__builtin_amdgcn_s_wait_tensorcnt)
#define FFN_USE_TDM 1
#endif
#endif
#ifndef FFN_USE_TDM
#define FFN_USE_TDM 0
#endif

#if FFN_USE_TDM
typedef __attribute__((ext_vector_type(4))) unsigned int tdm_u32x4;
typedef __attribute__((ext_vector_type(8))) int          tdm_i32x8;
typedef __attribute__((ext_vector_type(4))) int          tdm_i32x4;
#endif

// --------------------------- WMMA helpers ----------------------------------

__device__ __forceinline__ v8f wmma_f16(v16h a, v16h b, v8f c) {
  // D = A(16x32 f16) * B(32x16 f16) + C(16x16 f32)
  return __builtin_amdgcn_wmma_f32_16x16x32_f16(
      /*neg_a=*/false, a, /*neg_b=*/false, b,
      /*c_mod=*/(short)0, c, /*reuse_a=*/false, /*reuse_b=*/false);
}

// 16-bit A/B fragment layout (ISA 7.12.2): lane row = lane&15; per lane two
// contiguous 8-half chunks at K = kb and kb+16, kb = (lane>=16)*8.
__device__ __forceinline__ v16h load_frag_f16(const half_t* __restrict__ base,
                                              int ld, int k0) {
  int lane = threadIdx.x & 31;
  int r    = lane & 15;
  int kb   = k0 + ((lane >> 4) << 3);
  const half_t* p = base + (size_t)r * ld + kb;
  v16h f;
#pragma unroll
  for (int i = 0; i < 8; ++i) { f[i] = p[i]; f[8 + i] = p[16 + i]; }
  return f;
}

// Same fragment, fp32 source (weights) converted to f16 on the fly.
__device__ __forceinline__ v16h load_frag_f32cvt(const float* __restrict__ base,
                                                 int ld, int k0) {
  int lane = threadIdx.x & 31;
  int r    = lane & 15;
  int kb   = k0 + ((lane >> 4) << 3);
  const float* p = base + (size_t)r * ld + kb;
  v16h f;
#pragma unroll
  for (int i = 0; i < 8; ++i) {
    f[i]     = (_Float16)p[i];
    f[8 + i] = (_Float16)p[16 + i];
  }
  return f;
}

// B fragment with strided K (V in attention): elem(n,k) = base[n + k*ks]
__device__ __forceinline__ v16h load_frag_f16_kstride(const half_t* __restrict__ base,
                                                      int ks, int k0) {
  int lane = threadIdx.x & 31;
  int n    = lane & 15;
  int kb   = k0 + ((lane >> 4) << 3);
  v16h f;
#pragma unroll
  for (int i = 0; i < 8; ++i) {
    f[i]     = base[n + (size_t)(kb + i) * ks];
    f[8 + i] = base[n + (size_t)(kb + 16 + i) * ks];
  }
  return f;
}

// C/D 16x16 f32 tile: VGPR r -> row r + (lane>=16)*8, col = lane&15.
#define C_TILE_COORDS int c_col = (threadIdx.x & 15); int c_rb = ((threadIdx.x & 31) >> 4) << 3;

__device__ __forceinline__ float block_reduce_sum256(float v) {
  __shared__ float red[8];
  int lane = threadIdx.x & 31, wid = threadIdx.x >> 5;
#pragma unroll
  for (int off = 16; off; off >>= 1) v += __shfl_xor(v, off, 32);
  if (lane == 0) red[wid] = v;
  __syncthreads();
  float s = (threadIdx.x < 8) ? red[threadIdx.x] : 0.0f;
  if (wid == 0) {
#pragma unroll
    for (int off = 4; off; off >>= 1) s += __shfl_xor(s, off, 32);
    if (lane == 0) red[0] = s;
  }
  __syncthreads();
  return red[0];
}

// --------------------------- elementwise kernels ---------------------------

__global__ void k_rmsnorm_f16(const float* __restrict__ X,
                              const float* __restrict__ W,
                              half_t* __restrict__ OUT, int D) {
  int row = blockIdx.x;
  const float* xr = X + (size_t)row * D;
  float ss = 0.f;
  for (int i = threadIdx.x; i < D; i += 256) { float v = xr[i]; ss += v * v; }
  ss = block_reduce_sum256(ss);
  float sc = rsqrtf(ss / (float)D + EPSV);
  for (int i = threadIdx.x; i < D; i += 256)
    OUT[(size_t)row * D + i] = (_Float16)(xr[i] * sc * W[i]);
}

// rms over full row, then per-head RoPE, -> f16. W = row width (2048 or 1024)
__global__ void k_rmsrope_f16(const float* __restrict__ X,
                              const float* __restrict__ COS,
                              const float* __restrict__ SIN,
                              half_t* __restrict__ OUT, int W) {
  int s = blockIdx.x;
  const float* xr = X + (size_t)s * W;
  float ss = 0.f;
  for (int i = threadIdx.x; i < W; i += 256) { float v = xr[i]; ss += v * v; }
  ss = block_reduce_sum256(ss);
  float r = rsqrtf(ss / (float)W + EPSV);
  int pairs = W >> 1;
  for (int p = threadIdx.x; p < pairs; p += 256) {
    int i = p & 63;                     // pair index within head (DH/2 = 64)
    float c = COS[(size_t)s * 64 + i];
    float sn = SIN[(size_t)s * 64 + i];
    float x0 = xr[2 * p] * r, x1 = xr[2 * p + 1] * r;
    OUT[(size_t)s * W + 2 * p]     = (_Float16)(x0 * c - x1 * sn);
    OUT[(size_t)s * W + 2 * p + 1] = (_Float16)(x0 * sn + x1 * c);
  }
}

__global__ void k_cvt_f16(const float* __restrict__ X, half_t* __restrict__ OUT, int n) {
  for (int i = blockIdx.x * 256 + threadIdx.x; i < n; i += gridDim.x * 256)
    OUT[i] = (_Float16)X[i];
}

__global__ void k_final_sum(const float* __restrict__ H, const float* __restrict__ MOE,
                            const float* __restrict__ SH, float* __restrict__ OUT, int n) {
  for (int i = blockIdx.x * 256 + threadIdx.x; i < n; i += gridDim.x * 256)
    OUT[i] = H[i] + MOE[i] + SH[i];
}

// --------------------------- GEMM (C = A * B^T [+R]) -----------------------
// A: f16 [M,K]; B: fp32 [N,K] (weight, cvt on the fly). block 256 = 8 waves,
// block tile 64(M) x 128(N), wave tile 16x64.
template <bool RESID>
__global__ void k_gemm(const half_t* __restrict__ A, const float* __restrict__ B,
                       const float* __restrict__ R, float* __restrict__ C,
                       int M, int N, int K) {
  int wid = threadIdx.x >> 5;
  int wm = wid & 3, wn = wid >> 2;
  int m0 = blockIdx.y * 64 + wm * 16;
  int n0 = blockIdx.x * 128 + wn * 64;
  const half_t* Ab = A + (size_t)m0 * K;
  v8f acc[4];
#pragma unroll
  for (int j = 0; j < 4; ++j) acc[j] = (v8f){0,0,0,0,0,0,0,0};
  for (int k0 = 0; k0 < K; k0 += 32) {
    v16h a = load_frag_f16(Ab, K, k0);
#pragma unroll
    for (int j = 0; j < 4; ++j) {
      const float* Bj = B + (size_t)(n0 + 16 * j) * K;
      if ((k0 & 255) == 0) __builtin_prefetch(Bj + k0 + 256, 0, 1);
      v16h b = load_frag_f32cvt(Bj, K, k0);
      acc[j] = wmma_f16(a, b, acc[j]);
    }
  }
  C_TILE_COORDS
#pragma unroll
  for (int j = 0; j < 4; ++j) {
    float* Cb = C + (size_t)m0 * N + n0 + 16 * j;
    const float* Rb = RESID ? (R + (size_t)m0 * N + n0 + 16 * j) : nullptr;
#pragma unroll
    for (int r = 0; r < 8; ++r) {
      float v = acc[j][r];
      if (RESID) v += Rb[(size_t)(c_rb + r) * N + c_col];
      Cb[(size_t)(c_rb + r) * N + c_col] = v;
    }
  }
}

// --------------------------- attention -------------------------------------
// grid = (SEQ/16, NHQ), block 256 (8 waves).
// LDS: 16x2048 f32 scores + 16x2048 f16 probs + 16x128 f16 Q tile = 200704 B.
#define ATTN_LDS_BYTES (16 * SEQ * 4 + 16 * SEQ * 2 + 16 * DHEAD * 2)

__global__ void k_attn(const half_t* __restrict__ Q, const half_t* __restrict__ K,
                       const half_t* __restrict__ V, half_t* __restrict__ O) {
  extern __shared__ char smem[];
  float*  sc = (float*)smem;                        // [16][SEQ]
  half_t* pr = (half_t*)(smem + 16 * SEQ * 4);      // [16][SEQ]
  half_t* Qs = (half_t*)(smem + 16 * SEQ * 6);      // [16][DHEAD]
  int q0 = blockIdx.x * 16;
  int h  = blockIdx.y;
  int hk = h >> 1;                                  // GQA: n_rep = 2
  int wid = threadIdx.x >> 5, lane = threadIdx.x & 31;
  const float scale = 0.08838834764831845f;         // 1/sqrt(128)

  // phase 0: stage 16x128 Q tile to LDS via gfx1250 async copy (once, instead
  // of all 8 waves re-reading it from global).
  {
    int r = threadIdx.x >> 4;                       // 0..15
    int c = (threadIdx.x & 15) << 3;                // 0..120, 16B chunks
    unsigned dst = (unsigned)(uintptr_t)(Qs + r * DHEAD + c);
    unsigned long long src =
        (unsigned long long)(uintptr_t)(Q + (size_t)(q0 + r) * DMODEL + h * DHEAD + c);
    asm volatile("global_load_async_to_lds_b128 %0, %1, off"
                 :: "v"(dst), "v"(src) : "memory");
    asm volatile("s_wait_asynccnt 0x0" ::: "memory");
  }
  __syncthreads();

  v16h qa[4];
#pragma unroll
  for (int i = 0; i < 4; ++i) qa[i] = load_frag_f16(Qs, DHEAD, i * 32);

  // phase 1: scores for this wave's key slice [wid*256, +256)
  C_TILE_COORDS
  for (int j = 0; j < 16; ++j) {
    int kc = wid * 256 + j * 16;
    const half_t* Kb = K + (size_t)kc * KVDIM + hk * DHEAD;
    v8f acc = (v8f){0,0,0,0,0,0,0,0};
#pragma unroll
    for (int i = 0; i < 4; ++i)
      acc = wmma_f16(qa[i], load_frag_f16(Kb, KVDIM, i * 32), acc);
#pragma unroll
    for (int r = 0; r < 8; ++r) {
      int qr = c_rb + r;
      int kg = kc + c_col;
      float m = (kg <= q0 + qr) ? 0.0f : -1.0e9f;   // causal (start_pos = 0)
      sc[qr * SEQ + kg] = acc[r] * scale + m;
    }
  }
  __syncthreads();

  // phase 2: softmax; each wave owns 2 rows (wave32 shuffles).
  for (int rr = 0; rr < 2; ++rr) {
    int row = wid * 2 + rr;
    float mx = -3.4e38f;
    for (int k = lane; k < SEQ; k += 32) mx = fmaxf(mx, sc[row * SEQ + k]);
#pragma unroll
    for (int off = 16; off; off >>= 1) mx = fmaxf(mx, __shfl_xor(mx, off, 32));
    float sum = 0.f;
    for (int k = lane; k < SEQ; k += 32) sum += __expf(sc[row * SEQ + k] - mx);
#pragma unroll
    for (int off = 16; off; off >>= 1) sum += __shfl_xor(sum, off, 32);
    float inv = 1.0f / sum;
    for (int k = lane; k < SEQ; k += 32)
      pr[row * SEQ + k] = (_Float16)(__expf(sc[row * SEQ + k] - mx) * inv);
  }
  __syncthreads();

  // phase 3: out = P @ V; each wave owns one 16-col tile of DHEAD.
  const half_t* Vb = V + hk * DHEAD + wid * 16;
  v8f acc = (v8f){0,0,0,0,0,0,0,0};
  for (int kk = 0; kk < SEQ / 32; ++kk) {
    v16h a = load_frag_f16(pr, SEQ, kk * 32);
    v16h b = load_frag_f16_kstride(Vb, KVDIM, kk * 32);
    acc = wmma_f16(a, b, acc);
  }
#pragma unroll
  for (int r = 0; r < 8; ++r)
    O[(size_t)(q0 + c_rb + r) * DMODEL + h * DHEAD + wid * 16 + c_col] =
        (_Float16)acc[r];
}

// --------------------------- gating / routing ------------------------------

__global__ void k_gate(const half_t* __restrict__ Xn, const float* __restrict__ GW,
                       int* __restrict__ idx) {
  int wid = threadIdx.x >> 5, lane = threadIdx.x & 31;
  int t = blockIdx.x * 8 + wid;
  float best = -3.4e38f; int bi = 0;
  for (int e = 0; e < NEXP; ++e) {
    float a = 0.f;
    for (int k = lane; k < DMODEL; k += 32)
      a += (float)Xn[(size_t)t * DMODEL + k] * GW[(size_t)e * DMODEL + k];
#pragma unroll
    for (int off = 16; off; off >>= 1) a += __shfl_xor(a, off, 32);
    if (a > best) { best = a; bi = e; }   // first-max wins, like jnp.argmax
  }
  if (lane == 0) idx[t] = bi;
}

__global__ void k_route(const int* __restrict__ idx, int* counts, int* offs,
                        int* pos, int* perm) {
  if (threadIdx.x < NEXP) { counts[threadIdx.x] = 0; pos[threadIdx.x] = 0; }
  __syncthreads();
  for (int t = threadIdx.x; t < SEQ; t += 256) atomicAdd(&counts[idx[t]], 1);
  __syncthreads();
  if (threadIdx.x == 0) {
    int a = 0;
    for (int e = 0; e < NEXP; ++e) { offs[e] = a; a += counts[e]; }
  }
  __syncthreads();
  for (int t = threadIdx.x; t < SEQ; t += 256) {
    int e = idx[t];
    int p = atomicAdd(&pos[e], 1);
    perm[offs[e] + p] = t;
  }
}

// --------------------------- fused FFN tile --------------------------------
// 16 tokens x full D output. Loop DFFN in chunks of 256:
//   phase A: each wave computes 16x32 of t1/t3, silu-mul -> LDS f16
//   phase B: each wave accumulates its 16x256 slice of out (16 C tiles)
#define FFN_LDS_BYTES (16 * DMODEL * 2 + 16 * 256 * 2)

__device__ void ffn_tile(const half_t* __restrict__ Xn, const int* rows, int nv,
                         const float* __restrict__ W1, const float* __restrict__ W3,
                         const float* __restrict__ W2, float* __restrict__ OUT,
                         char* dsm) {
  half_t* Xt = (half_t*)dsm;                       // [16][DMODEL]
  half_t* Hm = (half_t*)(dsm + 16 * DMODEL * 2);   // [16][256]
  int tid = threadIdx.x, wid = tid >> 5;

#if FFN_USE_TDM
  // Stage gathered 16-token tile via the Tensor Data Mover in gather mode:
  // 16x 16-bit row indices, tile = 16 x 2048 f16 (D# layout per ISA 8.3-8.7).
  if (tid < 32) {
    unsigned lds = (unsigned)(uintptr_t)Xt;
    unsigned long long ga = (unsigned long long)(uintptr_t)Xn;
    tdm_u32x4 g0;
    g0[0] = 0x80000001u;                                   // gather_mode=1, 16b idx, count=1
    g0[1] = lds;                                           // lds_addr
    g0[2] = (unsigned)(ga & 0xffffffffu);                  // global_addr[31:0]
    g0[3] = (unsigned)((ga >> 32) & 0x01ffffffu) | (2u << 30); // addr[56:32] | type=2
    tdm_i32x8 g1;
    g1[0] = 0x00010000;                 // workgroup_mask=0, data_size=1 (2B)
    g1[1] = (int)(2048u << 16);         // tensor_dim0[15:0] (=DMODEL)
    g1[2] = (int)(2048u << 16);         // tensor_dim1[15:0] (=SEQ rows)
    g1[3] = (int)(2048u << 16);         // tile_dim0 = 2048
    g1[4] = 16;                         // tile_dim1 = #valid indices
    g1[5] = 2048;                       // tensor_dim0_stride lo
    g1[6] = 0;
    g1[7] = 0;
    tdm_i32x4 g2, g3;
#pragma unroll
    for (int i = 0; i < 4; ++i) {
      g2[i] = (rows[2 * i] & 0xffff) | (rows[2 * i + 1] << 16);
      g3[i] = (rows[8 + 2 * i] & 0xffff) | (rows[8 + 2 * i + 1] << 16);
    }
#if __clang_major__ >= 23
    tdm_i32x8 gz = {0, 0, 0, 0, 0, 0, 0, 0};
    __builtin_amdgcn_tensor_load_to_lds(g0, g1, g2, g3, gz, 0);
#else
    __builtin_amdgcn_tensor_load_to_lds(g0, g1, g2, g3, 0);
#endif
    __builtin_amdgcn_s_wait_tensorcnt((short)0);
  }
  __syncthreads();
#else
  // Fallback: gfx1250 async global->LDS copies.
  for (int c = tid; c < 16 * (DMODEL / 8); c += 256) {
    int r = c >> 8;
    int col = (c & 255) << 3;
    unsigned dst = (unsigned)(uintptr_t)(Xt + (size_t)r * DMODEL + col);
    unsigned long long src =
        (unsigned long long)(uintptr_t)(Xn + (size_t)rows[r] * DMODEL + col);
    asm volatile("global_load_async_to_lds_b128 %0, %1, off"
                 :: "v"(dst), "v"(src) : "memory");
  }
  asm volatile("s_wait_asynccnt 0x0" ::: "memory");
  __syncthreads();
#endif

  v8f accOut[16];
#pragma unroll
  for (int j = 0; j < 16; ++j) accOut[j] = (v8f){0,0,0,0,0,0,0,0};

  C_TILE_COORDS
  for (int f0 = 0; f0 < DFFN; f0 += 256) {
    // ---- phase A: t1/t3 for cols [f0 + wid*32, +32)
    v8f a1[2], a3[2];
#pragma unroll
    for (int j = 0; j < 2; ++j) { a1[j] = (v8f){0,0,0,0,0,0,0,0}; a3[j] = (v8f){0,0,0,0,0,0,0,0}; }
    const float* W1b = W1 + (size_t)(f0 + wid * 32) * DMODEL;
    const float* W3b = W3 + (size_t)(f0 + wid * 32) * DMODEL;
    for (int k0 = 0; k0 < DMODEL; k0 += 32) {
      v16h a = load_frag_f16(Xt, DMODEL, k0);
      if ((k0 & 255) == 0) {
        __builtin_prefetch(W1b + k0 + 256, 0, 1);
        __builtin_prefetch(W3b + k0 + 256, 0, 1);
      }
#pragma unroll
      for (int j = 0; j < 2; ++j) {
        a1[j] = wmma_f16(a, load_frag_f32cvt(W1b + (size_t)(16 * j) * DMODEL, DMODEL, k0), a1[j]);
        a3[j] = wmma_f16(a, load_frag_f32cvt(W3b + (size_t)(16 * j) * DMODEL, DMODEL, k0), a3[j]);
      }
    }
#pragma unroll
    for (int j = 0; j < 2; ++j)
#pragma unroll
      for (int r = 0; r < 8; ++r) {
        float t1 = a1[j][r], t3 = a3[j][r];
        float s = t1 / (1.0f + __expf(-t1));       // silu
        Hm[(size_t)(c_rb + r) * 256 + wid * 32 + j * 16 + c_col] = (_Float16)(s * t3);
      }
    __syncthreads();

    // ---- phase B: out[:, wid*256 .. +256) += Hm @ W2_chunk^T
    const float* W2b = W2 + (size_t)(wid * 256) * DFFN + f0;
    for (int k0 = 0; k0 < 256; k0 += 32) {
      v16h a = load_frag_f16(Hm, 256, k0);
#pragma unroll
      for (int j = 0; j < 16; ++j) {
        accOut[j] = wmma_f16(a, load_frag_f32cvt(W2b + (size_t)(16 * j) * DFFN, DFFN, k0), accOut[j]);
      }
    }
    __syncthreads();   // Hm reused next chunk
  }

  // store (scatter rows; skip padded rows)
#pragma unroll
  for (int j = 0; j < 16; ++j) {
    int n0 = wid * 256 + j * 16;
#pragma unroll
    for (int r = 0; r < 8; ++r) {
      int rl = c_rb + r;
      if (rl < nv)
        OUT[(size_t)rows[rl] * DMODEL + n0 + c_col] = accOut[j][r];
    }
  }
}

// MoE: grid covers worst-case tile count (<= 128 + 7). Each block resolves
// its (expert, tile) from device-side counts.
__global__ void k_moe(const half_t* __restrict__ Xn,
                      const float* __restrict__ w1, const float* __restrict__ w3,
                      const float* __restrict__ w2,
                      const int* __restrict__ counts, const int* __restrict__ offs,
                      const int* __restrict__ perm, float* __restrict__ OUT) {
  extern __shared__ char dsm[];
  int rem = blockIdx.x, e = -1, t = 0;
  for (int ee = 0; ee < NEXP; ++ee) {
    int nt = (counts[ee] + 15) >> 4;
    if (rem < nt) { e = ee; t = rem; break; }
    rem -= nt;
  }
  if (e < 0) return;
  __shared__ int rows[16];
  if (threadIdx.x < 16) {
    int i = t * 16 + threadIdx.x;
    rows[threadIdx.x] = (i < counts[e]) ? perm[offs[e] + i] : 0;
  }
  __syncthreads();
  int nv = counts[e] - t * 16; if (nv > 16) nv = 16;
  ffn_tile(Xn, rows, nv,
           w1 + (size_t)e * DFFN * DMODEL,
           w3 + (size_t)e * DFFN * DMODEL,
           w2 + (size_t)e * DMODEL * DFFN, OUT, dsm);
}

__global__ void k_sffn(const half_t* __restrict__ Xn,
                       const float* __restrict__ sw1, const float* __restrict__ sw3,
                       const float* __restrict__ sw2, float* __restrict__ OUT) {
  extern __shared__ char dsm[];
  __shared__ int rows[16];
  if (threadIdx.x < 16) rows[threadIdx.x] = blockIdx.x * 16 + threadIdx.x;
  __syncthreads();
  ffn_tile(Xn, rows, 16, sw1, sw3, sw2, OUT, dsm);
}

// --------------------------- host launch -----------------------------------

extern "C" void kernel_launch(void* const* d_in, const int* in_sizes, int n_in,
                              void* d_out, int out_size, void* d_ws, size_t ws_size,
                              hipStream_t stream) {
  const float* x      = (const float*)d_in[0];
  const float* fcos   = (const float*)d_in[2];
  const float* fsin   = (const float*)d_in[3];
  const float* anw    = (const float*)d_in[5];
  const float* wq     = (const float*)d_in[6];
  const float* wk     = (const float*)d_in[7];
  const float* wv     = (const float*)d_in[8];
  const float* wo     = (const float*)d_in[9];
  const float* fnw    = (const float*)d_in[10];
  const float* gatew  = (const float*)d_in[11];
  const float* w1     = (const float*)d_in[12];
  const float* w2     = (const float*)d_in[13];
  const float* w3     = (const float*)d_in[14];
  const float* sw1    = (const float*)d_in[15];
  const float* sw2    = (const float*)d_in[16];
  const float* sw3    = (const float*)d_in[17];
  float* out = (float*)d_out;

  char* w = (char*)d_ws;
  size_t off = 0;
  auto alloc = [&](size_t bytes) -> void* {
    void* p = w + off;
    off += (bytes + 255) & ~(size_t)255;
    return p;
  };
  half_t* hn16   = (half_t*)alloc((size_t)SEQ * DMODEL * 2);
  float*  xq     = (float*) alloc((size_t)SEQ * DMODEL * 4);
  float*  xk     = (float*) alloc((size_t)SEQ * KVDIM * 4);
  float*  xv     = (float*) alloc((size_t)SEQ * KVDIM * 4);
  half_t* q16    = (half_t*)alloc((size_t)SEQ * DMODEL * 2);
  half_t* k16    = (half_t*)alloc((size_t)SEQ * KVDIM * 2);
  half_t* v16    = (half_t*)alloc((size_t)SEQ * KVDIM * 2);
  half_t* at16   = (half_t*)alloc((size_t)SEQ * DMODEL * 2);
  float*  hbuf   = (float*) alloc((size_t)SEQ * DMODEL * 4);
  half_t* hn2    = (half_t*)alloc((size_t)SEQ * DMODEL * 2);
  float*  moe    = (float*) alloc((size_t)SEQ * DMODEL * 4);
  float*  shb    = (float*) alloc((size_t)SEQ * DMODEL * 4);
  int*    idx    = (int*)   alloc(SEQ * 4);
  int*    counts = (int*)   alloc(NEXP * 4);
  int*    offsb  = (int*)   alloc(NEXP * 4);
  int*    pos    = (int*)   alloc(NEXP * 4);
  int*    perm   = (int*)   alloc(SEQ * 4);
  (void)in_sizes; (void)n_in; (void)out_size; (void)ws_size;

  // 1. attention RMSNorm -> f16
  k_rmsnorm_f16<<<SEQ, 256, 0, stream>>>(x, anw, hn16, DMODEL);
  // 2. QKV projections
  k_gemm<false><<<dim3(DMODEL/128, SEQ/64), 256, 0, stream>>>(hn16, wq, nullptr, xq, SEQ, DMODEL, DMODEL);
  k_gemm<false><<<dim3(KVDIM/128,  SEQ/64), 256, 0, stream>>>(hn16, wk, nullptr, xk, SEQ, KVDIM, DMODEL);
  k_gemm<false><<<dim3(KVDIM/128,  SEQ/64), 256, 0, stream>>>(hn16, wv, nullptr, xv, SEQ, KVDIM, DMODEL);
  // 3. QK-RMS + RoPE -> f16 ; V -> f16
  k_rmsrope_f16<<<SEQ, 256, 0, stream>>>(xq, fcos, fsin, q16, DMODEL);
  k_rmsrope_f16<<<SEQ, 256, 0, stream>>>(xk, fcos, fsin, k16, KVDIM);
  k_cvt_f16<<<2048, 256, 0, stream>>>(xv, v16, SEQ * KVDIM);
  // 4. attention
  k_attn<<<dim3(SEQ/16, NHQ), 256, ATTN_LDS_BYTES, stream>>>(q16, k16, v16, at16);
  // 5. output projection + residual
  k_gemm<true><<<dim3(DMODEL/128, SEQ/64), 256, 0, stream>>>(at16, wo, x, hbuf, SEQ, DMODEL, DMODEL);
  // 6. FFN RMSNorm -> f16
  k_rmsnorm_f16<<<SEQ, 256, 0, stream>>>(hbuf, fnw, hn2, DMODEL);
  // 7. gate + routing
  k_gate<<<SEQ/8, 256, 0, stream>>>(hn2, gatew, idx);
  k_route<<<1, 256, 0, stream>>>(idx, counts, offsb, pos, perm);
  // 8. MoE (grouped) + shared expert
  k_moe <<<SEQ/16 + NEXP, 256, FFN_LDS_BYTES, stream>>>(hn2, w1, w3, w2, counts, offsb, perm, moe);
  k_sffn<<<SEQ/16,        256, FFN_LDS_BYTES, stream>>>(hn2, sw1, sw3, sw2, shb);
  // 9. final residual
  k_final_sum<<<4096, 256, 0, stream>>>(hbuf, moe, shb, out, SEQ * DMODEL);
}
